// TMix_x060c2_10591389352193
// MI455X (gfx1250) — matrix operand order repaired and checked by reading, not compiled, not run on hardware
//
#include <hip/hip_runtime.h>
#include <hip/hip_bf16.h>

#define Bn   8
#define Tn   2048
#define Cn   2048
#define Hn   32
#define HSn  64
#define DMIX 32
#define DDEC 64

typedef __bf16 bf16;
typedef __attribute__((ext_vector_type(16))) __bf16 v16bf;
typedef __attribute__((ext_vector_type(8)))  float  v8f;
typedef unsigned int u32;

// ---------------------------------------------------------------------------
// Weight convert + transpose: W[c_in][c_out] fp32 -> Wt[c_out][c_in] bf16
// ---------------------------------------------------------------------------
__global__ __launch_bounds__(256) void convert_transpose(
    const float* __restrict__ W, bf16* __restrict__ Wt)
{
  __shared__ float tile[32][33];
  const int tx = threadIdx.x & 31;
  const int ty = threadIdx.x >> 5;            // 0..7
  const long c0 = (long)blockIdx.x * 32;      // src col (c_out)
  const long r0 = (long)blockIdx.y * 32;      // src row (c_in)
#pragma unroll
  for (int i = 0; i < 4; ++i)
    tile[ty + i * 8][tx] = W[(r0 + ty + i * 8) * Cn + c0 + tx];
  __syncthreads();
#pragma unroll
  for (int i = 0; i < 4; ++i)
    Wt[(c0 + ty + i * 8) * Cn + r0 + tx] = (bf16)tile[tx][ty + i * 8];
}

// ---------------------------------------------------------------------------
// Fused token-shift + 5-way LoRA mixing + decay/value2 LoRA.
// One block per (b,t) row of C=2048.
// ---------------------------------------------------------------------------
__global__ __launch_bounds__(256) void mix_kernel(
    const float* __restrict__ x, const float* __restrict__ shift_state,
    const float* __restrict__ maa_x, const float* __restrict__ maa_r,
    const float* __restrict__ maa_k, const float* __restrict__ maa_v,
    const float* __restrict__ maa_w, const float* __restrict__ maa_v2,
    const float* __restrict__ w1,    const float* __restrict__ w2,
    const float* __restrict__ tdecay,const float* __restrict__ dec_w1,
    const float* __restrict__ dec_w2,const float* __restrict__ v2_w1,
    const float* __restrict__ v2_w2,
    bf16* __restrict__ oxr, bf16* __restrict__ oxk, bf16* __restrict__ oxv,
    bf16* __restrict__ oxv2,
    float* __restrict__ od, float* __restrict__ ov2a,
    float* __restrict__ oxlast)
{
  __shared__ float sx[Cn], sdx[Cn], sA[Cn], sB[Cn];
  __shared__ float th[5 * DMIX], th2[DDEC], th3[DDEC];
  const int tid = threadIdx.x;
  const int b = blockIdx.x / Tn;
  const int t = blockIdx.x % Tn;
  const size_t row = (size_t)blockIdx.x * Cn;    // blockIdx.x == b*Tn + t
  const float* xr_ = x + row;
  const float* pr_ = (t == 0) ? (shift_state + (size_t)b * Cn) : (x + row - Cn);

  for (int c = tid; c < Cn; c += 256) {
    float xv = xr_[c];
    float dx = pr_[c] - xv;
    sx[c] = xv; sdx[c] = dx;
    sA[c] = xv + dx * maa_x[c];                  // xxx
  }
  __syncthreads();

  if (tid < 5 * DMIX) {                          // hidden = tanh(xxx @ w1)
    float acc = 0.f;
#pragma unroll 8
    for (int c = 0; c < Cn; ++c) acc = fmaf(sA[c], w1[(size_t)c * 160 + tid], acc);
    th[tid] = tanhf(acc);
  }
  __syncthreads();

  for (int i = 0; i < Cn / 256; ++i) {
    int c = tid + i * 256;
    float mg[5];
#pragma unroll
    for (int g = 0; g < 5; ++g) {
      float a = 0.f;
      const float* wp = w2 + ((size_t)g * DMIX) * Cn + c;
#pragma unroll 8
      for (int dd = 0; dd < DMIX; ++dd) a = fmaf(th[g * DMIX + dd], wp[(size_t)dd * Cn], a);
      mg[g] = a;
    }
    float xv = sx[c], dx = sdx[c];
    oxr[row + c] = (bf16)(xv + dx * (maa_r[c] + mg[0]));
    oxk[row + c] = (bf16)(xv + dx * (maa_k[c] + mg[1]));
    oxv[row + c] = (bf16)(xv + dx * (maa_v[c] + mg[2]));
    float xw  = xv + dx * (maa_w[c]  + mg[3]);
    float xv2 = xv + dx * (maa_v2[c] + mg[4]);
    sA[c] = xw;
    sB[c] = xv2;
    oxv2[row + c] = (bf16)xv2;
  }
  __syncthreads();

  if (tid < DDEC) {                              // tanh(xw @ dec_w1)
    float acc = 0.f;
#pragma unroll 8
    for (int c = 0; c < Cn; ++c) acc = fmaf(sA[c], dec_w1[(size_t)c * DDEC + tid], acc);
    th2[tid] = tanhf(acc);
  } else if (tid < 2 * DDEC) {                   // tanh(xv2 @ v2_w1)
    int j = tid - DDEC;
    float acc = 0.f;
#pragma unroll 8
    for (int c = 0; c < Cn; ++c) acc = fmaf(sB[c], v2_w1[(size_t)c * DDEC + j], acc);
    th3[j] = tanhf(acc);
  }
  __syncthreads();

  for (int i = 0; i < Cn / 256; ++i) {
    int c = tid + i * 256;
    float wv = tdecay[c];
    float va = 0.f;
#pragma unroll 8
    for (int j = 0; j < DDEC; ++j) {
      wv = fmaf(th2[j], dec_w2[(size_t)j * Cn + c], wv);
      va = fmaf(th3[j], v2_w2[(size_t)j * Cn + c], va);
    }
    od[row + c]   = __expf(-__expf(wv));         // per-channel decay
    ov2a[row + c] = va;                          // v2 lora addend
  }
  if (t == Tn - 1)
    for (int c = tid; c < Cn; c += 256) oxlast[(size_t)b * Cn + c] = sx[c];
}

// ---------------------------------------------------------------------------
// bf16 WMMA GEMM: Out[M,N] = A[M,K] * Bt[N,K]^T  (Bt is B pre-transposed)
// Block tile 128x128x32, 8 waves (4x2), each wave 2x4 of 16x16 WMMA tiles.
// ---------------------------------------------------------------------------
__global__ __launch_bounds__(256) void gemm_bf16_nt(
    const bf16* __restrict__ A, const bf16* __restrict__ Bt,
    float* __restrict__ Out, int M, int N, int K)
{
  __shared__ u32 sa[128 * 17];                   // [m][16 dwords + 1 pad]
  __shared__ u32 sb[128 * 17];                   // [n][16 dwords + 1 pad]
  const int tid  = threadIdx.x;
  const int lane = tid & 31;
  const int wave = tid >> 5;
  const int wm   = wave >> 1;                    // 0..3 : 32-row band
  const int wn   = wave & 1;                     // 0..1 : 64-col band
  const int half = lane >> 4;                    // 0/1
  const int l16  = lane & 15;
  const long m0 = (long)blockIdx.x * 128;
  const long n0 = (long)blockIdx.y * 128;

  v8f acc[2][4];
#pragma unroll
  for (int mi = 0; mi < 2; ++mi)
#pragma unroll
    for (int ni = 0; ni < 4; ++ni) acc[mi][ni] = (v8f){0, 0, 0, 0, 0, 0, 0, 0};

  const int lr = tid >> 1;                       // 0..127 : tile row
  const int lh = tid & 1;                        // 16-element chunk
  const bf16* aptr = A  + (m0 + lr) * (long)K + lh * 16;
  const bf16* bptr = Bt + (n0 + lr) * (long)K + lh * 16;

  for (int k0 = 0; k0 < K; k0 += 32) {
    uint4 av0 = ((const uint4*)(aptr + k0))[0];
    uint4 av1 = ((const uint4*)(aptr + k0))[1];
    uint4 bv0 = ((const uint4*)(bptr + k0))[0];
    uint4 bv1 = ((const uint4*)(bptr + k0))[1];
    __syncthreads();
    u32* pa = sa + lr * 17 + lh * 8;
    pa[0] = av0.x; pa[1] = av0.y; pa[2] = av0.z; pa[3] = av0.w;
    pa[4] = av1.x; pa[5] = av1.y; pa[6] = av1.z; pa[7] = av1.w;
    u32* pb = sb + lr * 17 + lh * 8;
    pb[0] = bv0.x; pb[1] = bv0.y; pb[2] = bv0.z; pb[3] = bv0.w;
    pb[4] = bv1.x; pb[5] = bv1.y; pb[6] = bv1.z; pb[7] = bv1.w;
    __syncthreads();

    union UV { u32 u[8]; v16bf v; };
    UV af[2], bfr[4];
#pragma unroll
    for (int mi = 0; mi < 2; ++mi) {             // A frag: 16x32, ISA layout
      const u32* p = sa + (wm * 32 + mi * 16 + l16) * 17;
#pragma unroll
      for (int v = 0; v < 4; ++v) {
        af[mi].u[v]     = p[half * 4 + v];       // K = half*8 + 2v (+1)
        af[mi].u[4 + v] = p[8 + half * 4 + v];   // K = 16 + half*8 + 2v (+1)
      }
    }
#pragma unroll
    for (int ni = 0; ni < 4; ++ni) {             // B frag: 32x16, ISA layout
      const u32* p = sb + (wn * 64 + ni * 16 + l16) * 17;
#pragma unroll
      for (int v = 0; v < 4; ++v) {
        bfr[ni].u[v]     = p[half * 4 + v];      // K = half*8 + 2v (+1)
        bfr[ni].u[4 + v] = p[8 + half * 4 + v];  // K = 16 + half*8 + 2v (+1)
      }
    }
#pragma unroll
    for (int mi = 0; mi < 2; ++mi)
#pragma unroll
      for (int ni = 0; ni < 4; ++ni)
        acc[mi][ni] = __builtin_amdgcn_wmma_f32_16x16x32_bf16(
            false, af[mi].v, false, bfr[ni].v, (short)0, acc[mi][ni], false, false);
  }

#pragma unroll
  for (int mi = 0; mi < 2; ++mi)
#pragma unroll
    for (int ni = 0; ni < 4; ++ni) {
      long col = n0 + wn * 64 + ni * 16 + l16;
#pragma unroll
      for (int v = 0; v < 8; ++v) {
        long rowi = m0 + wm * 32 + mi * 16 + half * 8 + v;   // C/D layout
        Out[rowi * (long)N + col] = acc[mi][ni][v];
      }
    }
}

// ---------------------------------------------------------------------------
// WKV linear recurrence: one block per (b,h); thread j owns S[:,j] in VGPRs.
// y_t = r_t . S  (u==0, so pre-update state);  S = d (.) S + (k*(1-d)) v^T
// ---------------------------------------------------------------------------
__global__ __launch_bounds__(64) void scan_kernel(
    const float* __restrict__ r, const float* __restrict__ k,
    const float* __restrict__ v, const float* __restrict__ d,
    const float* __restrict__ S0, float* __restrict__ y0,
    float* __restrict__ Sout)
{
  const int bh = blockIdx.x;
  const int b = bh / Hn, h = bh % Hn;
  const int j = threadIdx.x;
  float S[HSn];
  const size_t sbase = ((size_t)bh * HSn) * HSn + j;
#pragma unroll
  for (int kk = 0; kk < HSn; ++kk) S[kk] = S0[sbase + (size_t)kk * HSn];
  __shared__ float sr[HSn], sk[HSn], sv[HSn], sd[HSn];
  size_t off = ((size_t)b * Tn) * Cn + (size_t)h * HSn + j;
  for (int t = 0; t < Tn; ++t, off += Cn) {
    float dj = d[off];
    sr[j] = r[off];
    sd[j] = dj;
    sk[j] = k[off] * (1.0f - dj);
    sv[j] = v[off];
    __syncthreads();
    float vj = sv[j];
    float y = 0.f;
#pragma unroll
    for (int kk = 0; kk < HSn; ++kk) {
      y = fmaf(sr[kk], S[kk], y);
      S[kk] = fmaf(sd[kk], S[kk], sk[kk] * vj);
    }
    y0[off] = y;
    __syncthreads();
  }
#pragma unroll
  for (int kk = 0; kk < HSn; ++kk) Sout[sbase + (size_t)kk * HSn] = S[kk];
}

// ---------------------------------------------------------------------------
// y = LN(y0 + v2gemm + v2lora) -> bf16 (input to final Wo GEMM)
// ---------------------------------------------------------------------------
__global__ __launch_bounds__(256) void ln_kernel(
    const float* __restrict__ y0, const float* __restrict__ v2g,
    const float* __restrict__ v2a, const float* __restrict__ lnw,
    const float* __restrict__ lnb, bf16* __restrict__ oy)
{
  __shared__ float red[256];
  const int tid = threadIdx.x;
  const size_t row = (size_t)blockIdx.x * Cn;
  float vals[Cn / 256];
  float s = 0.f;
#pragma unroll
  for (int i = 0; i < Cn / 256; ++i) {
    int c = tid + i * 256;
    float yv = y0[row + c] + v2g[row + c] + v2a[row + c];
    vals[i] = yv; s += yv;
  }
  red[tid] = s; __syncthreads();
  for (int st = 128; st > 0; st >>= 1) { if (tid < st) red[tid] += red[tid + st]; __syncthreads(); }
  float mu = red[0] * (1.0f / Cn);
  __syncthreads();
  s = 0.f;
#pragma unroll
  for (int i = 0; i < Cn / 256; ++i) { float dv = vals[i] - mu; s = fmaf(dv, dv, s); }
  red[tid] = s; __syncthreads();
  for (int st = 128; st > 0; st >>= 1) { if (tid < st) red[tid] += red[tid + st]; __syncthreads(); }
  float rstd = rsqrtf(red[0] * (1.0f / Cn) + 1e-5f);
#pragma unroll
  for (int i = 0; i < Cn / 256; ++i) {
    int c = tid + i * 256;
    oy[row + c] = (bf16)((vals[i] - mu) * rstd * lnw[c] + lnb[c]);
  }
}

// ---------------------------------------------------------------------------
extern "C" void kernel_launch(void* const* d_in, const int* in_sizes, int n_in,
                              void* d_out, int out_size, void* d_ws, size_t ws_size,
                              hipStream_t stream) {
  const float* x      = (const float*)d_in[0];
  const float* shift  = (const float*)d_in[1];
  const float* wkv0   = (const float*)d_in[2];
  const float* maa_x  = (const float*)d_in[3];
  const float* maa_r  = (const float*)d_in[4];
  const float* maa_k  = (const float*)d_in[5];
  const float* maa_v  = (const float*)d_in[6];
  const float* maa_w  = (const float*)d_in[7];
  const float* maa_v2 = (const float*)d_in[8];
  const float* w1     = (const float*)d_in[9];
  const float* w2     = (const float*)d_in[10];
  const float* tdecay = (const float*)d_in[11];
  const float* dec_w1 = (const float*)d_in[12];
  const float* dec_w2 = (const float*)d_in[13];
  const float* v2_w1  = (const float*)d_in[14];
  const float* v2_w2  = (const float*)d_in[15];
  // d_in[16] = time_faaaa : unused (u == 0 in this variant)
  const float* Wr     = (const float*)d_in[17];
  const float* Wk     = (const float*)d_in[18];
  const float* Wv     = (const float*)d_in[19];
  const float* Wo     = (const float*)d_in[20];
  const float* ln_w   = (const float*)d_in[21];
  const float* ln_b   = (const float*)d_in[22];

  const size_t NC = (size_t)Bn * Tn * Cn;        // 33,554,432
  const size_t CC = (size_t)Cn * Cn;

  // workspace layout
  bf16*  xr_b  = (bf16*)d_ws;
  bf16*  xk_b  = xr_b + NC;
  bf16*  xv_b  = xk_b + NC;
  bf16*  xv2_b = xv_b + NC;
  float* dbuf  = (float*)(xv2_b + NC);
  float* v2a   = dbuf + NC;
  float* rb    = v2a + NC;
  float* kb    = rb + NC;
  float* vb    = kb + NC;
  float* v2g   = vb + NC;
  float* y0    = v2g + NC;
  bf16*  yln   = (bf16*)(y0 + NC);
  bf16*  wrt   = yln + NC;
  bf16*  wkt   = wrt + CC;
  bf16*  wvt   = wkt + CC;
  bf16*  wot   = wvt + CC;

  float* out_y     = (float*)d_out;
  float* out_S     = out_y + NC;
  float* out_xlast = out_S + (size_t)Bn * Hn * HSn * HSn;

  // 1. weight convert + transpose (fp32 -> bf16, [c_out][c_in])
  dim3 tgrid(Cn / 32, Cn / 32);
  convert_transpose<<<tgrid, 256, 0, stream>>>(Wr, wrt);
  convert_transpose<<<tgrid, 256, 0, stream>>>(Wk, wkt);
  convert_transpose<<<tgrid, 256, 0, stream>>>(Wv, wvt);
  convert_transpose<<<tgrid, 256, 0, stream>>>(Wo, wot);

  // 2. fused token-shift + LoRA mixing
  mix_kernel<<<Bn * Tn, 256, 0, stream>>>(
      x, shift, maa_x, maa_r, maa_k, maa_v, maa_w, maa_v2,
      w1, w2, tdecay, dec_w1, dec_w2, v2_w1, v2_w2,
      xr_b, xk_b, xv_b, xv2_b, dbuf, v2a, out_xlast);

  // 3. r/k/v/v2 projections via bf16 WMMA
  dim3 ggrid(Bn * Tn / 128, Cn / 128);
  gemm_bf16_nt<<<ggrid, 256, 0, stream>>>(xr_b,  wrt, rb,  Bn * Tn, Cn, Cn);
  gemm_bf16_nt<<<ggrid, 256, 0, stream>>>(xk_b,  wkt, kb,  Bn * Tn, Cn, Cn);
  gemm_bf16_nt<<<ggrid, 256, 0, stream>>>(xv_b,  wvt, vb,  Bn * Tn, Cn, Cn);
  gemm_bf16_nt<<<ggrid, 256, 0, stream>>>(xv2_b, wvt, v2g, Bn * Tn, Cn, Cn);

  // 4. WKV recurrent scan; also emits S_final
  scan_kernel<<<Bn * Hn, 64, 0, stream>>>(rb, kb, vb, dbuf, wkv0, y0, out_S);

  // 5. y + v2 paths, LayerNorm -> bf16
  ln_kernel<<<Bn * Tn, 256, 0, stream>>>(y0, v2g, v2a, ln_w, ln_b, yln);

  // 6. output projection via bf16 WMMA -> d_out[0:B*T*C]
  gemm_bf16_nt<<<ggrid, 256, 0, stream>>>(yln, wot, out_y, Bn * Tn, Cn, Cn);

  (void)in_sizes; (void)n_in; (void)out_size; (void)ws_size;
}
